// GATLinkPred_56624848830744
// MI455X (gfx1250) — matrix operand order repaired and checked by reading, not compiled, not run on hardware
//
#include <hip/hip_runtime.h>

// ---------------------------------------------------------------------------
// Types for CDNA5 WMMA (wave32): 16x16x32 bf16 -> f32 accumulate
// ---------------------------------------------------------------------------
typedef __attribute__((ext_vector_type(16))) __bf16 bf16x16;
typedef __attribute__((ext_vector_type(8)))  __bf16 bf16x8;
typedef __attribute__((ext_vector_type(8)))  float  f32x8;

// Order-preserving float -> u32 encoding so segment-max can use atomicMax(u32).
// memset(0) is below enc(-inf), so zero-init is the identity element.
__device__ __forceinline__ unsigned encf(float x) {
  union { float f; unsigned u; } v; v.f = x;
  return (v.u & 0x80000000u) ? ~v.u : (v.u | 0x80000000u);
}
__device__ __forceinline__ float decf(unsigned e) {
  union { unsigned u; float f; } v;
  v.u = (e & 0x80000000u) ? (e ^ 0x80000000u) : ~e;
  return v.f;
}

// ---------------------------------------------------------------------------
// Weight pre-pack: W[K x 128] f32  ->  fragment-ready bf16
// layout: [ntile(8)][ktile(K/32)][lane(32)][j(16)]
//   element j of lane L in tile (nt,kt) is W[kt*32 + (L<16?0:8) + (j&7) + ((j&8)?16:0)][nt*16 + (L&15)]
// Runs once per GEMM (<=16K elements); every consumer wave then loads its whole
// B fragment with two contiguous 16B loads.
// ---------------------------------------------------------------------------
__global__ __launch_bounds__(256) void k_pack_w(
    const float* __restrict__ W, __bf16* __restrict__ Wp, int K)
{
  int i = blockIdx.x * 256 + threadIdx.x;
  int total = 8 * (K / 32) * 512;
  if (i >= total) return;
  int j    = i & 15;
  int lane = (i >> 4) & 31;
  int tile = i >> 9;
  int kt   = tile % (K / 32);
  int nt   = tile / (K / 32);
  int n = nt * 16 + (lane & 15);
  int k = kt * 32 + ((lane < 16) ? 0 : 8) + (j & 7) + ((j & 8) ? 16 : 0);
  Wp[i] = (__bf16)W[(size_t)k * 128 + n];
}

// ---------------------------------------------------------------------------
// GEMM: out[N x 128] = A[N x K] @ W[K x 128] + bias (+ optional emb gather)
// Block = 256 threads = 8 waves; block owns 32 rows, wave w owns cols 16w..16w+15
// for BOTH 16-row tiles: one B fragment feeds two WMMAs per k-step (2x math per
// B fetch).  A staged f32->bf16 into padded LDS (row stride K+8 elems -> the
// 16B ds_load_b128 fragment reads hit rotating bank groups, conflict-free).
// K compile-time: MAC loop fully unrolls into 2*(K/32) chained WMMAs.
// Requires N % 16 == 0 (holds: 100000, 30000); last block may have one tile.
// ---------------------------------------------------------------------------
template<int K, bool ADD>
__global__ __launch_bounds__(256) void k_gemm_bf16(
    const float* __restrict__ A, const __bf16* __restrict__ Wp,
    const float* __restrict__ bias,
    const float* __restrict__ addend, const int* __restrict__ addid,
    float* __restrict__ out, int N)
{
  constexpr int LDSK = K + 8;
  __shared__ __align__(32) __bf16 As[32 * LDSK];
  const int row0 = blockIdx.x * 32;
  const int lane = threadIdx.x & 31;
  const int wave = threadIdx.x >> 5;
  const bool has2 = (row0 + 16) < N;       // second 16-row tile valid?

  for (int idx = threadIdx.x; idx < 32 * K; idx += 256) {
    int r = idx / K, c = idx - r * K;
    int rr = row0 + r; if (rr >= N) rr = N - 1;   // clamp (reads only)
    As[r * LDSK + c] = (__bf16)A[(size_t)rr * K + c];
  }
  __syncthreads();

  // A-frag (16-bit A 16x32): lanes 0-15 row=lane, K in {0..7,16..23};
  // lanes 16-31 row=lane-16, K in {8..15,24..31}.
  const int mrow = lane & 15;
  const int kb1  = (lane < 16) ? 0 : 8;
  const __bf16* arow0 = &As[mrow * LDSK];
  const __bf16* arow1 = &As[(mrow + 16) * LDSK];
  const __bf16* bp    = Wp + ((size_t)(wave * (K / 32)) * 32 + lane) * 16;

  f32x8 acc0 = {}, acc1 = {};
#pragma unroll
  for (int kt = 0; kt < K / 32; ++kt) {
    bf16x16 b = *(const bf16x16*)(bp + (size_t)kt * 512);
    bf16x8 a0lo = *(const bf16x8*)(arow0 + kt * 32 + kb1);
    bf16x8 a0hi = *(const bf16x8*)(arow0 + kt * 32 + kb1 + 16);
    bf16x8 a1lo = *(const bf16x8*)(arow1 + kt * 32 + kb1);
    bf16x8 a1hi = *(const bf16x8*)(arow1 + kt * 32 + kb1 + 16);
    bf16x16 a0 = __builtin_shufflevector(a0lo, a0hi,
        0, 1, 2, 3, 4, 5, 6, 7, 8, 9, 10, 11, 12, 13, 14, 15);
    bf16x16 a1 = __builtin_shufflevector(a1lo, a1hi,
        0, 1, 2, 3, 4, 5, 6, 7, 8, 9, 10, 11, 12, 13, 14, 15);
    acc0 = __builtin_amdgcn_wmma_f32_16x16x32_bf16(
        false, a0, false, b, (short)0, acc0, false, false);
    acc1 = __builtin_amdgcn_wmma_f32_16x16x32_bf16(
        false, a1, false, b, (short)0, acc1, false, false);
  }

  // C/D layout: VGPR r -> lanes 0-15 M=r, lanes 16-31 M=8+r.
  const int   ncol  = wave * 16 + (lane & 15);
  const int   mbase = row0 + ((lane < 16) ? 0 : 8);
  const float bv    = bias[ncol];
#pragma unroll
  for (int r = 0; r < 8; ++r) {
    int m = mbase + r;
    float v = acc0[r] + bv;
    if (ADD) v += addend[(size_t)addid[m] * 128 + ncol];
    out[(size_t)m * 128 + ncol] = v;
  }
  if (has2) {
#pragma unroll
    for (int r = 0; r < 8; ++r) {
      int m = mbase + 16 + r;
      float v = acc1[r] + bv;
      if (ADD) v += addend[(size_t)addid[m] * 128 + ncol];
      out[(size_t)m * 128 + ncol] = v;
    }
  }
}

// ---------------------------------------------------------------------------
// Edge phase.  One wave per edge; lane owns 4 of 128 features, head = lane/8
// (D = 32 per head).  att flat index == lane*4+j.
// ---------------------------------------------------------------------------
__global__ __launch_bounds__(256) void k_edge_logits(
    const float* __restrict__ xl, const float* __restrict__ xr,
    const int* __restrict__ src, const int* __restrict__ dst,
    const float* __restrict__ att, float* __restrict__ logits,
    unsigned* __restrict__ amax, int E)
{
  int e = blockIdx.x * 8 + (threadIdx.x >> 5);
  if (e >= E) return;
  int lane = threadIdx.x & 31;
  int s = src[e], d = dst[e];
  const float4 va = *(const float4*)(xl + (size_t)s * 128 + lane * 4);
  const float4 vb = *(const float4*)(xr + (size_t)d * 128 + lane * 4);
  const float4 wa = *(const float4*)(att + lane * 4);
  float p = 0.f, m;
  m = va.x + vb.x; m = (m > 0.f) ? m : 0.2f * m; p += m * wa.x;
  m = va.y + vb.y; m = (m > 0.f) ? m : 0.2f * m; p += m * wa.y;
  m = va.z + vb.z; m = (m > 0.f) ? m : 0.2f * m; p += m * wa.z;
  m = va.w + vb.w; m = (m > 0.f) ? m : 0.2f * m; p += m * wa.w;
  p += __shfl_xor(p, 1, 32);
  p += __shfl_xor(p, 2, 32);
  p += __shfl_xor(p, 4, 32);
  if ((lane & 7) == 0) {
    int h = lane >> 3;
    logits[(size_t)e * 4 + h] = p;
    atomicMax(&amax[(size_t)d * 4 + h], encf(p));
  }
}

__global__ __launch_bounds__(256) void k_edge_exp(
    const int* __restrict__ dst, const unsigned* __restrict__ amax,
    float* __restrict__ logits, float* __restrict__ denom, int EH)
{
  int i = blockIdx.x * 256 + threadIdx.x;
  if (i >= EH) return;
  int e = i >> 2, h = i & 3;
  int d = dst[e];
  float v = __expf(logits[i] - decf(amax[(size_t)d * 4 + h]));
  logits[i] = v;
  atomicAdd(&denom[(size_t)d * 4 + h], v);
}

__global__ __launch_bounds__(256) void k_edge_agg(
    const float* __restrict__ xl,
    const int* __restrict__ src, const int* __restrict__ dst,
    const float* __restrict__ eweight, const float* __restrict__ denom,
    float* __restrict__ accum, int E)
{
  int e = blockIdx.x * 8 + (threadIdx.x >> 5);
  if (e >= E) return;
  int lane = threadIdx.x & 31;
  int s = src[e], d = dst[e];
  int h = lane >> 3;
  float alpha = eweight[(size_t)e * 4 + h] / (denom[(size_t)d * 4 + h] + 1e-16f);
  const float4 va = *(const float4*)(xl + (size_t)s * 128 + lane * 4);
  float* o = accum + (size_t)d * 128 + lane * 4;
  atomicAdd(o + 0, alpha * va.x);
  atomicAdd(o + 1, alpha * va.y);
  atomicAdd(o + 2, alpha * va.z);
  atomicAdd(o + 3, alpha * va.w);
}

__global__ __launch_bounds__(256) void k_finalize(
    float* __restrict__ accum, const float* __restrict__ bo, int total, int relu)
{
  int i = blockIdx.x * 256 + threadIdx.x;
  if (i >= total) return;
  float v = accum[i] + bo[i & 127];
  accum[i] = relu ? fmaxf(v, 0.f) : v;
}

__global__ __launch_bounds__(256) void k_classify(
    const float* __restrict__ xu, const float* __restrict__ xp,
    const int* __restrict__ es, const int* __restrict__ ed,
    float* __restrict__ out, int EL)
{
  int e = blockIdx.x * 8 + (threadIdx.x >> 5);
  if (e >= EL) return;
  int lane = threadIdx.x & 31;
  const float4 a = *(const float4*)(xu + (size_t)es[e] * 128 + lane * 4);
  const float4 b = *(const float4*)(xp + (size_t)ed[e] * 128 + lane * 4);
  float p = a.x * b.x + a.y * b.y + a.z * b.z + a.w * b.w;
  p += __shfl_xor(p, 1, 32);
  p += __shfl_xor(p, 2, 32);
  p += __shfl_xor(p, 4, 32);
  p += __shfl_xor(p, 8, 32);
  p += __shfl_xor(p, 16, 32);
  if (lane == 0) out[e] = p;
}

// ---------------------------------------------------------------------------
// Host orchestration (graph-capture safe: kernel launches + MemsetAsync only)
// ---------------------------------------------------------------------------
extern "C" void kernel_launch(void* const* d_in, const int* in_sizes, int n_in,
                              void* d_out, int out_size, void* d_ws, size_t ws_size,
                              hipStream_t stream)
{
  const float* x_user   = (const float*)d_in[0];
  const float* x_prod   = (const float*)d_in[1];
  const float* Wu       = (const float*)d_in[2];
  const float* bu       = (const float*)d_in[3];
  const float* Wp       = (const float*)d_in[4];
  const float* bp       = (const float*)d_in[5];
  const float* user_emb = (const float*)d_in[6];
  const float* prod_emb = (const float*)d_in[7];
  const float* Wl       = (const float*)d_in[8];
  const float* bl       = (const float*)d_in[9];
  const float* Wr       = (const float*)d_in[10];
  const float* br       = (const float*)d_in[11];
  const float* att      = (const float*)d_in[12];
  const float* bo       = (const float*)d_in[13];
  const int* uid        = (const int*)d_in[14];
  const int* pid        = (const int*)d_in[15];
  const int* buy_src    = (const int*)d_in[16];
  const int* buy_dst    = (const int*)d_in[17];
  const int* el_src     = (const int*)d_in[18];
  const int* el_dst     = (const int*)d_in[19];

  const int NU = in_sizes[14];
  const int NP = in_sizes[15];
  const int E  = in_sizes[16];
  const int EL = in_sizes[18];
  const int FU = in_sizes[0] / NU;
  const int FP = in_sizes[1] / NP;
  const int C = 128, H = 4;

  float* p   = (float*)d_ws;
  float* xu  = p; p += (size_t)NU * C;
  float* xp  = p; p += (size_t)NP * C;
  float* nu  = p; p += (size_t)NU * C;
  float* np_ = p; p += (size_t)NP * C;
  float* xlb = p; p += (size_t)NU * C;       // src-transformed features
  float* xrb = p; p += (size_t)NU * C;       // dst-transformed features
  float* elog = p; p += (size_t)E * H;       // logits, then exp weights
  unsigned* amax = (unsigned*)p; p += (size_t)NU * H;
  float* denom = p; p += (size_t)NU * H;
  __bf16* wpack = (__bf16*)p;                // 16384 bf16 = 32KB

  auto gemm = [&](const float* A, const float* W, const float* bias,
                  const float* add, const int* id, float* out, int N, int K) {
    int total = 8 * (K / 32) * 512;
    k_pack_w<<<(total + 255) / 256, 256, 0, stream>>>(W, wpack, K);
    dim3 g((N + 31) / 32), b(256);
    if (K == 32) {
      if (add) k_gemm_bf16<32, true ><<<g, b, 0, stream>>>(A, wpack, bias, add, id, out, N);
      else     k_gemm_bf16<32, false><<<g, b, 0, stream>>>(A, wpack, bias, nullptr, nullptr, out, N);
    } else if (K == 64) {
      if (add) k_gemm_bf16<64, true ><<<g, b, 0, stream>>>(A, wpack, bias, add, id, out, N);
      else     k_gemm_bf16<64, false><<<g, b, 0, stream>>>(A, wpack, bias, nullptr, nullptr, out, N);
    } else {
      if (add) k_gemm_bf16<128, true ><<<g, b, 0, stream>>>(A, wpack, bias, add, id, out, N);
      else     k_gemm_bf16<128, false><<<g, b, 0, stream>>>(A, wpack, bias, nullptr, nullptr, out, N);
    }
  };

  // Encoders: xu = x_user@Wu + bu + user_emb[uid];  xp likewise.
  gemm(x_user, Wu, bu, user_emb, uid, xu, NU, FU);
  gemm(x_prod, Wp, bp, prod_emb, pid, xp, NP, FP);

  float* cu = xu; float* cp = xp; float* ou = nu; float* op = np_;

  for (int l = 0; l < 3; ++l) {
    for (int dir = 0; dir < 2; ++dir) {
      const int idx = l * 2 + dir;
      const float* Wlp  = Wl + (size_t)idx * C * C;
      const float* blp  = bl + (size_t)idx * C;
      const float* Wrp  = Wr + (size_t)idx * C * C;
      const float* brp  = br + (size_t)idx * C;
      const float* attp = att + (size_t)idx * C;
      const float* bop  = bo + (size_t)idx * C;

      const float *xsrc, *xdst; const int *esrc, *edst;
      int Nsrc, Ndst; float* acc;
      if (dir == 0) { xsrc = cu; xdst = cp; esrc = buy_src; edst = buy_dst;
                      Nsrc = NU; Ndst = NP; acc = op; }
      else          { xsrc = cp; xdst = cu; esrc = buy_dst; edst = buy_src;
                      Nsrc = NP; Ndst = NU; acc = ou; }

      gemm(xsrc, Wlp, blp, nullptr, nullptr, xlb, Nsrc, C);
      gemm(xdst, Wrp, brp, nullptr, nullptr, xrb, Ndst, C);

      hipMemsetAsync(amax,  0, (size_t)Ndst * H * sizeof(unsigned), stream);
      hipMemsetAsync(denom, 0, (size_t)Ndst * H * sizeof(float), stream);
      hipMemsetAsync(acc,   0, (size_t)Ndst * C * sizeof(float), stream);

      const int eb = (E + 7) / 8;
      k_edge_logits<<<eb, 256, 0, stream>>>(xlb, xrb, esrc, edst, attp, elog, amax, E);
      k_edge_exp<<<(E * H + 255) / 256, 256, 0, stream>>>(edst, amax, elog, denom, E * H);
      k_edge_agg<<<eb, 256, 0, stream>>>(xlb, esrc, edst, elog, denom, acc, E);
      k_finalize<<<((size_t)Ndst * C + 255) / 256, 256, 0, stream>>>(acc, bop, Ndst * C, (l < 2) ? 1 : 0);
    }
    float* t;
    t = cu; cu = ou; ou = t;
    t = cp; cp = op; op = t;
  }

  k_classify<<<(EL + 7) / 8, 256, 0, stream>>>(cu, cp, el_src, el_dst, (float*)d_out, EL);
}
